// SparseMOE_34351148433722
// MI455X (gfx1250) — matrix operand order repaired and checked by reading, not compiled, not run on hardware
//
#include <hip/hip_runtime.h>

#define T_TOK 8192
#define H_DIM 1024
#define E_NUM 8
#define KPR   (H_DIM / 2)   // 512 packed bf16 k-pairs per row

typedef __attribute__((ext_vector_type(16))) __bf16   v16bf;
typedef __attribute__((ext_vector_type(8)))  float    v8f;
typedef __attribute__((ext_vector_type(4)))  unsigned v4u;
typedef __attribute__((ext_vector_type(8)))  int      v8i;
typedef __attribute__((ext_vector_type(4)))  int      v4i;

union Frag16 { v16bf v; uint4 q[2]; };

__device__ __forceinline__ unsigned short f2bf(float f) {
    unsigned a = __builtin_bit_cast(unsigned, f);
    return (unsigned short)((a + 0x7FFFu + ((a >> 16) & 1u)) >> 16);
}
__device__ __forceinline__ unsigned f2bf_pk(float lo, float hi) {
    return (unsigned)f2bf(lo) | ((unsigned)f2bf(hi) << 16);
}

// ---- convert x (f32) -> k-pair-packed bf16 rows: xbf[t][kp] ----
__global__ __launch_bounds__(256) void conv_x(const float* __restrict__ x,
                                              unsigned* __restrict__ xbf) {
    int idx = blockIdx.x * 256 + threadIdx.x;        // one uint4 (4 dwords = 8 bf16)
    const float4* xin = (const float4*)x;
    float4 a = xin[idx * 2], b = xin[idx * 2 + 1];
    uint4 o;
    o.x = f2bf_pk(a.x, a.y); o.y = f2bf_pk(a.z, a.w);
    o.z = f2bf_pk(b.x, b.y); o.w = f2bf_pk(b.z, b.w);
    ((uint4*)xbf)[idx] = o;
}

// ---- convert+transpose expert_w (f32 [e][k][n]) -> wbf[e][n][kp] packed bf16 ----
__global__ __launch_bounds__(256) void conv_w(const float* __restrict__ ew,
                                              unsigned* __restrict__ wbf) {
    __shared__ unsigned short Tl[128 * 33];
    const int e = blockIdx.z, k0 = blockIdx.x * 128, n0 = blockIdx.y * 32;
    const int tid = threadIdx.x;
    const int nc = tid & 31, kr = tid >> 5;          // read: n coalesced
    const float* src = ew + (size_t)e * H_DIM * H_DIM;
#pragma unroll
    for (int r = 0; r < 16; ++r) {
        int k = kr + r * 8;
        Tl[k * 33 + nc] = f2bf(src[(size_t)(k0 + k) * H_DIM + n0 + nc]);
    }
    __syncthreads();
    const int kpl = tid & 63, nl0 = tid >> 6;        // write: kp coalesced
    unsigned* dst = wbf + (size_t)e * H_DIM * KPR;
#pragma unroll
    for (int i = 0; i < 8; ++i) {
        int nl = nl0 + i * 4;
        unsigned d = (unsigned)Tl[(2 * kpl) * 33 + nl] |
                     ((unsigned)Tl[(2 * kpl + 1) * 33 + nl] << 16);
        dst[(size_t)(n0 + nl) * KPR + (k0 >> 1) + kpl] = d;
    }
}

// ---------------- Router: 1 wave32 per token ----------------
__global__ __launch_bounds__(256) void moe_router(
    const float* __restrict__ x, const float* __restrict__ gw,
    const float* __restrict__ gb, float* __restrict__ rlog,
    int* __restrict__ counts, int* __restrict__ elist, float* __restrict__ eprob)
{
    const int lane = threadIdx.x & 31;
    const int wave = threadIdx.x >> 5;
    const int t = blockIdx.x * 8 + wave;

    float acc[E_NUM];
#pragma unroll
    for (int e = 0; e < E_NUM; ++e) acc[e] = 0.f;
    const float* xr = x + (size_t)t * H_DIM;
#pragma unroll 4
    for (int i = 0; i < H_DIM / 32; ++i) {
        int h = i * 32 + lane;
        float xv = xr[h];
        const float4* g = (const float4*)(gw + (size_t)h * E_NUM);
        float4 g0 = g[0], g1 = g[1];
        acc[0] += xv * g0.x; acc[1] += xv * g0.y; acc[2] += xv * g0.z; acc[3] += xv * g0.w;
        acc[4] += xv * g1.x; acc[5] += xv * g1.y; acc[6] += xv * g1.z; acc[7] += xv * g1.w;
    }
#pragma unroll
    for (int e = 0; e < E_NUM; ++e)
#pragma unroll
        for (int off = 16; off > 0; off >>= 1)
            acc[e] += __shfl_xor(acc[e], off, 32);

    if (lane == 0) {
        float logit[E_NUM], pr[E_NUM];
        float mx = -1e30f;
#pragma unroll
        for (int e = 0; e < E_NUM; ++e) { logit[e] = acc[e] + gb[e]; mx = fmaxf(mx, logit[e]); }
        float sum = 0.f;
#pragma unroll
        for (int e = 0; e < E_NUM; ++e) { pr[e] = __expf(logit[e] - mx); sum += pr[e]; }
        float inv = 1.f / sum;
#pragma unroll
        for (int e = 0; e < E_NUM; ++e) { pr[e] *= inv; rlog[(size_t)t * E_NUM + e] = logit[e]; }
        int i0 = 0;
#pragma unroll
        for (int e = 1; e < E_NUM; ++e) if (pr[e] > pr[i0]) i0 = e;
        int i1 = (i0 == 0) ? 1 : 0;
#pragma unroll
        for (int e = 0; e < E_NUM; ++e) if (e != i0 && pr[e] > pr[i1]) i1 = e;
        float p0 = pr[i0], p1 = pr[i1];
        float rn = 1.f / (p0 + p1);
        p0 *= rn; p1 *= rn;
        int pos0 = atomicAdd(&counts[i0], 1);
        elist[i0 * T_TOK + pos0] = t; eprob[i0 * T_TOK + pos0] = p0;
        int pos1 = atomicAdd(&counts[i1], 1);
        elist[i1 * T_TOK + pos1] = t; eprob[i1 * T_TOK + pos1] = p1;
    }
}

// ------------- Expert GEMM: 128x128 tile, bf16 WMMA; B tile via TDM -------------
__global__ __launch_bounds__(256) void moe_expert_gemm(
    const unsigned* __restrict__ xbf, const unsigned* __restrict__ wbf,
    const float* __restrict__ eb, const int* __restrict__ counts,
    const int* __restrict__ elist, const float* __restrict__ eprob,
    float* __restrict__ out)
{
    const int e   = blockIdx.z;
    const int cnt = counts[e];
    const int bm  = blockIdx.x * 128;
    if (bm >= cnt) return;
    const int n0  = blockIdx.y * 128;

    const int tid  = threadIdx.x;
    const int lane = tid & 31, wave = tid >> 5;
    const int wm = wave & 3, wn = wave >> 2;   // 8 waves: 4(M) x 2(N), 32x64 each
    const int hh = lane >> 4, nn = lane & 15;

    __shared__ unsigned Alds[128 * 20];        // 16 data dwords + 4 pad per row
    __shared__ unsigned Blds[128 * 20];        // TDM pads 16->20 dwords per row

    const int*   lst = elist + e * T_TOK;
    const float* prb = eprob + e * T_TOK;

    const int lm = tid >> 1, seg = (tid & 1) * 8;
    const int arow = bm + lm;
    const int atok = (arow < cnt) ? lst[arow] : 0;
    const unsigned* axp = xbf + (size_t)atok * KPR + seg;

    const unsigned lds_b = (unsigned)(size_t)&Blds[0];     // low 32 bits = LDS offset
    const unsigned* wtile0 = wbf + (size_t)e * H_DIM * KPR + (size_t)n0 * KPR;

    v8f acc[2][4];
#pragma unroll
    for (int mi = 0; mi < 2; ++mi)
#pragma unroll
        for (int ni = 0; ni < 4; ++ni)
#pragma unroll
            for (int v = 0; v < 8; ++v) acc[mi][ni][v] = 0.f;

    for (int kb = 0; kb < H_DIM; kb += 32) {
        const uint4* ap = (const uint4*)(axp + (kb >> 1));
        uint4 a0 = ap[0], a1 = ap[1];

        __syncthreads();                       // previous iteration's LDS reads done

        if (wave == 0) {
            // TDM: 128 rows x 16 dwords, stride 512 dwords, pad 4 per 16 -> 20-dword rows
            unsigned long long ga = (unsigned long long)(size_t)(wtile0 + (kb >> 1));
            v4u g0; v8i g1; v4i g2; v4i g3; v8i g4;
            g0[0] = 1u;
            g0[1] = lds_b;
            g0[2] = (unsigned)ga;
            g0[3] = ((unsigned)(ga >> 32) & 0x1FFFFFFu) | (2u << 30);
            g1[0] = (int)((2u << 16) | (1u << 20) | (3u << 22) | (3u << 25));
            g1[1] = (int)((unsigned)(KPR & 0xFFFF) << 16);
            g1[2] = (int)((unsigned)(H_DIM & 0xFFFF) << 16);
            g1[3] = (int)(16u << 16);
            g1[4] = 128;
            g1[5] = KPR;
            g1[6] = 0; g1[7] = 0;
            g2[0] = g2[1] = g2[2] = g2[3] = 0;
            g3[0] = g3[1] = g3[2] = g3[3] = 0;
            g4[0] = g4[1] = g4[2] = g4[3] = 0;
            g4[4] = g4[5] = g4[6] = g4[7] = 0;
            __builtin_amdgcn_tensor_load_to_lds(g0, g1, g2, g3, g4, 0);
        }

        unsigned* arp = &Alds[lm * 20 + seg];
        *(uint4*)(arp)     = a0;
        *(uint4*)(arp + 4) = a1;

        if (wave == 0) __builtin_amdgcn_s_wait_tensorcnt(0);
        __syncthreads();

        Frag16 bfr[4];
#pragma unroll
        for (int ni = 0; ni < 4; ++ni) {
            const unsigned* bp = &Blds[(wn * 64 + ni * 16 + nn) * 20 + hh * 4];
            bfr[ni].q[0] = *(const uint4*)(bp);
            bfr[ni].q[1] = *(const uint4*)(bp + 8);
        }
#pragma unroll
        for (int mi = 0; mi < 2; ++mi) {
            Frag16 afr;
            const unsigned* apL = &Alds[(wm * 32 + mi * 16 + nn) * 20 + hh * 4];
            afr.q[0] = *(const uint4*)(apL);
            afr.q[1] = *(const uint4*)(apL + 8);
#pragma unroll
            for (int ni = 0; ni < 4; ++ni)
                acc[mi][ni] = __builtin_amdgcn_wmma_f32_16x16x32_bf16(
                    false, afr.v, false, bfr[ni].v, (short)0, acc[mi][ni], false, false);
        }
    }

    float bcol[4];
#pragma unroll
    for (int ni = 0; ni < 4; ++ni)
        bcol[ni] = eb[(size_t)e * H_DIM + n0 + wn * 64 + ni * 16 + nn];
#pragma unroll
    for (int mi = 0; mi < 2; ++mi) {
#pragma unroll
        for (int v = 0; v < 8; ++v) {
            int r = bm + wm * 32 + mi * 16 + hh * 8 + v;
            if (r < cnt) {
                int t   = lst[r];
                float p = prb[r];
                float* ob = out + (size_t)t * H_DIM + n0 + wn * 64 + nn;
#pragma unroll
                for (int ni = 0; ni < 4; ++ni)
                    unsafeAtomicAdd(ob + ni * 16, p * (acc[mi][ni][v] + bcol[ni]));
            }
        }
    }
}

extern "C" void kernel_launch(void* const* d_in, const int* in_sizes, int n_in,
                              void* d_out, int out_size, void* d_ws, size_t ws_size,
                              hipStream_t stream) {
    const float* x  = (const float*)d_in[0];
    const float* gw = (const float*)d_in[1];
    const float* gb = (const float*)d_in[2];
    const float* ew = (const float*)d_in[3];
    const float* eb = (const float*)d_in[4];
    float* out  = (float*)d_out;
    float* rlog = out + (size_t)T_TOK * H_DIM;

    char* ws = (char*)d_ws;
    int*      counts = (int*)ws;
    int*      elist  = (int*)(ws + 256);
    float*    eprob  = (float*)(ws + 256 + (size_t)E_NUM * T_TOK * 4);
    unsigned* xbf    = (unsigned*)(ws + (1u << 20));
    unsigned* wbf    = (unsigned*)(ws + (1u << 20) + ((size_t)T_TOK * KPR * 4));

    (void)hipMemsetAsync(d_out, 0, (size_t)T_TOK * H_DIM * sizeof(float), stream);
    (void)hipMemsetAsync(d_ws, 0, 256, stream);

    conv_x<<<(T_TOK * KPR) / (4 * 256), 256, 0, stream>>>(x, xbf);
    conv_w<<<dim3(H_DIM / 128, H_DIM / 32, E_NUM), 256, 0, stream>>>(ew, wbf);
    moe_router<<<T_TOK / 8, 256, 0, stream>>>(x, gw, gb, rlog, counts, elist, eprob);

    dim3 grid(64, 8, 8);
    moe_expert_gemm<<<grid, 256, 0, stream>>>(xbf, wbf, eb, counts, elist, eprob, out);
}
